// GlobalAttention_42004780155299
// MI455X (gfx1250) — matrix operand order repaired and checked by reading, not compiled, not run on hardware
//
#include <hip/hip_runtime.h>

#define Bsz 32
#define Ssz 2048
#define Hsz 1024

#define BLOCK 512        // 16 waves : 4 S-tiles x 4 N-tiles
#define ROWS  64         // S-rows per workgroup
#define NCH   64         // N columns per chunk (4 N-waves x 16)
#define KCH   32         // K per WMMA step

typedef __bf16 bf16x16 __attribute__((ext_vector_type(16)));
typedef float  f32x8   __attribute__((ext_vector_type(8)));

struct alignas(16) u128 { unsigned int x0, x1, x2, x3; };
struct alignas(8)  u64x { unsigned int a, b; };
union Frag { u128 q[2]; bf16x16 v; };

__device__ __forceinline__ unsigned int f2bf_rne(float f) {
    unsigned int u = __float_as_uint(f);
    u += 0x7FFFu + ((u >> 16) & 1u);      // round-to-nearest-even to bf16
    return u >> 16;
}
__device__ __forceinline__ float bf2f(unsigned int h) {
    return __uint_as_float(h << 16);
}
__device__ __forceinline__ void split_bf16(float v, unsigned int& h, unsigned int& l) {
    h = f2bf_rne(v);
    l = f2bf_rne(v - bf2f(h));
}

// ---------------------------------------------------------------------------
// Kernel 0: transpose+split Ua_w (f32, k-major) -> UaT_hi/lo (bf16, n-major)
// ---------------------------------------------------------------------------
__global__ void __launch_bounds__(256)
uaconv_kernel(const float* __restrict__ Ua_w,
              unsigned short* __restrict__ UaT_hi,
              unsigned short* __restrict__ UaT_lo) {
    __shared__ float tile[32][33];
    const int n0 = blockIdx.x * 32;
    const int k0 = blockIdx.y * 32;
    const int c  = threadIdx.x & 31;
    const int r0 = threadIdx.x >> 5;          // 8 rows per pass
    #pragma unroll
    for (int r = r0; r < 32; r += 8)
        tile[r][c] = Ua_w[(size_t)(k0 + r) * Hsz + (n0 + c)];   // tile[kk][nn]
    __syncthreads();
    #pragma unroll
    for (int r = r0; r < 32; r += 8) {        // write row n = n0+r, col k = k0+c
        unsigned int h, l;
        split_bf16(tile[c][r], h, l);         // element (k=k0+c, n=n0+r)
        const size_t o = (size_t)(n0 + r) * Hsz + (k0 + c);
        UaT_hi[o] = (unsigned short)h;
        UaT_lo[o] = (unsigned short)l;
    }
}

// ---------------------------------------------------------------------------
// Kernel 1: qkb[b][h] = query[b]·Wa_w[:,h] + Wa_b[h] + Ua_b[h]
// ---------------------------------------------------------------------------
__global__ void __launch_bounds__(256)
qproj_kernel(const float* __restrict__ query, const float* __restrict__ Wa_w,
             const float* __restrict__ Wa_b, const float* __restrict__ Ua_b,
             float* __restrict__ qkb) {
    __shared__ float qs[Hsz];
    const int b = blockIdx.y;
    const int h = blockIdx.x * 256 + threadIdx.x;
    for (int k = threadIdx.x; k < Hsz; k += 256) qs[k] = query[b * Hsz + k];
    __syncthreads();
    float acc = 0.f;
    const float* wcol = Wa_w + h;
    #pragma unroll 8
    for (int k = 0; k < Hsz; ++k) acc = fmaf(qs[k], wcol[(size_t)k * Hsz], acc);
    qkb[b * Hsz + h] = acc + Wa_b[h] + Ua_b[h];
}

// ---------------------------------------------------------------------------
// Kernel 2 (WMMA): fused  scores[b][s] = Va · tanh(qkb[b] + keys[b,s,:]@Ua) + Va_b
// bf16x3 split precision: A·B ~= Ah·Bh + Ah·Bl + Al·Bh  (f32 accumulate)
// A (keys tile) staged in LDS; B fragments read straight from L2-resident UaT.
// ---------------------------------------------------------------------------
__global__ void __launch_bounds__(BLOCK)
scores_kernel(const float* __restrict__ keys,
              const unsigned short* __restrict__ UaT_hi,
              const unsigned short* __restrict__ UaT_lo,
              const float* __restrict__ qkb,  const float* __restrict__ Va_w,
              const float* __restrict__ Va_b, float* __restrict__ scores) {
    extern __shared__ char smem[];
    unsigned short* Ahi = (unsigned short*)smem;          // ROWS*Hsz bf16
    unsigned short* Alo = Ahi + ROWS * Hsz;               // ROWS*Hsz bf16
    float* scoreAcc     = (float*)(Alo + ROWS * Hsz);     // ROWS floats

    const int b     = blockIdx.y;
    const int sBase = blockIdx.x * ROWS;
    const int tid   = threadIdx.x;
    const int lane  = tid & 31;
    const int wave  = tid >> 5;
    const int waveS = wave >> 2;   // 0..3 : S-tile
    const int waveN = wave & 3;    // 0..3 : N-tile

    // ---- stage keys tile as bf16 hi/lo planes (float4 loads, b64 stores) --
    const float4* kv = (const float4*)(keys + ((size_t)b * Ssz + sBase) * Hsz);
    u64x* aHv = (u64x*)Ahi;
    u64x* aLv = (u64x*)Alo;
    for (int i = tid; i < ROWS * Hsz / 4; i += BLOCK) {
        float4 v = kv[i];
        unsigned int h0, l0, h1, l1, h2, l2, h3, l3;
        split_bf16(v.x, h0, l0); split_bf16(v.y, h1, l1);
        split_bf16(v.z, h2, l2); split_bf16(v.w, h3, l3);
        u64x hp; hp.a = h0 | (h1 << 16); hp.b = h2 | (h3 << 16);
        u64x lp; lp.a = l0 | (l1 << 16); lp.b = l2 | (l3 << 16);
        aHv[i] = hp;
        aLv[i] = lp;
    }
    if (tid < ROWS) scoreAcc[tid] = 0.f;
    __syncthreads();

    const int halfLane = lane & 15;
    const int mRow  = waveS * 16 + halfLane;   // A-fragment row (all 32 lanes)
    const int offA0 = (lane < 16) ? 0 : 8;     // A K-interleave per ISA table
    const int offB0 = (lane < 16) ? 0 : 8;     // B K-interleave per ISA table

    for (int n0 = 0; n0 < Hsz; n0 += NCH) {
        const int nGlob = n0 + waveN * 16 + halfLane;   // this lane's column
        const unsigned short* gbh = UaT_hi + (size_t)nGlob * Hsz + offB0;
        const unsigned short* gbl = UaT_lo + (size_t)nGlob * Hsz + offB0;
        const unsigned short* ar  = Ahi + (size_t)mRow * Hsz + offA0;
        const unsigned short* al  = Alo + (size_t)mRow * Hsz + offA0;

        f32x8 acc = {};
        for (int k0 = 0; k0 < Hsz; k0 += KCH) {
            Frag aH, aL, bH, bL;
            aH.q[0] = *(const u128*)(ar + k0);
            aH.q[1] = *(const u128*)(ar + k0 + 16);
            aL.q[0] = *(const u128*)(al + k0);
            aL.q[1] = *(const u128*)(al + k0 + 16);
            bH.q[0] = *(const u128*)(gbh + k0);
            bH.q[1] = *(const u128*)(gbh + k0 + 16);
            bL.q[0] = *(const u128*)(gbl + k0);
            bL.q[1] = *(const u128*)(gbl + k0 + 16);

            acc = __builtin_amdgcn_wmma_f32_16x16x32_bf16(false, aH.v, false, bH.v,
                                                          (short)0, acc, false, false);
            acc = __builtin_amdgcn_wmma_f32_16x16x32_bf16(false, aH.v, false, bL.v,
                                                          (short)0, acc, false, false);
            acc = __builtin_amdgcn_wmma_f32_16x16x32_bf16(false, aL.v, false, bH.v,
                                                          (short)0, acc, false, false);
        }

        // ---- epilogue: tanh + Va dot, reduce over the 16 columns --------
        const float bias = qkb[b * Hsz + nGlob];
        const float va   = Va_w[nGlob];
        float psum[8];
        #pragma unroll
        for (int r = 0; r < 8; ++r) psum[r] = tanhf(acc[r] + bias) * va;
        #pragma unroll
        for (int off = 1; off < 16; off <<= 1) {
            #pragma unroll
            for (int r = 0; r < 8; ++r) psum[r] += __shfl_xor(psum[r], off, 32);
        }
        if (halfLane == 0) {
            const int mBase = waveS * 16 + ((lane >= 16) ? 8 : 0);
            #pragma unroll
            for (int r = 0; r < 8; ++r) atomicAdd(&scoreAcc[mBase + r], psum[r]);
        }
    }
    __syncthreads();
    if (tid < ROWS) scores[b * Ssz + sBase + tid] = scoreAcc[tid] + Va_b[0];
}

// ---------------------------------------------------------------------------
// Kernel 3: softmax over S per batch
// ---------------------------------------------------------------------------
__global__ void __launch_bounds__(256)
softmax_kernel(const float* __restrict__ scores, float* __restrict__ weights) {
    __shared__ float red[256];
    const int b = blockIdx.x, tid = threadIdx.x;
    float m = -3.402823e38f;
    for (int s = tid; s < Ssz; s += 256) m = fmaxf(m, scores[b * Ssz + s]);
    red[tid] = m; __syncthreads();
    for (int o = 128; o > 0; o >>= 1) {
        if (tid < o) red[tid] = fmaxf(red[tid], red[tid + o]);
        __syncthreads();
    }
    const float M = red[0]; __syncthreads();
    float sum = 0.f;
    for (int s = tid; s < Ssz; s += 256) sum += __expf(scores[b * Ssz + s] - M);
    red[tid] = sum; __syncthreads();
    for (int o = 128; o > 0; o >>= 1) {
        if (tid < o) red[tid] += red[tid + o];
        __syncthreads();
    }
    const float inv = 1.0f / red[0];
    for (int s = tid; s < Ssz; s += 256)
        weights[b * Ssz + s] = __expf(scores[b * Ssz + s] - M) * inv;
}

// ---------------------------------------------------------------------------
// Kernel 4: context[b][h] = sum_s weights[b][s] * keys[b][s][h]
// ---------------------------------------------------------------------------
__global__ void __launch_bounds__(256)
context_kernel(const float* __restrict__ weights, const float* __restrict__ keys,
               float* __restrict__ context) {
    __shared__ float w[Ssz];
    const int b = blockIdx.y;
    const int h = blockIdx.x * 256 + threadIdx.x;
    for (int s = threadIdx.x; s < Ssz; s += 256) w[s] = weights[b * Ssz + s];
    __syncthreads();
    float acc = 0.f;
    const float* kb = keys + (size_t)b * Ssz * Hsz + h;
    #pragma unroll 4
    for (int s = 0; s < Ssz; ++s) acc = fmaf(w[s], kb[(size_t)s * Hsz], acc);
    context[b * Hsz + h] = acc;
}

// ---------------------------------------------------------------------------
extern "C" void kernel_launch(void* const* d_in, const int* in_sizes, int n_in,
                              void* d_out, int out_size, void* d_ws, size_t ws_size,
                              hipStream_t stream) {
    const float* query = (const float*)d_in[0];
    const float* keys  = (const float*)d_in[1];
    const float* Wa_w  = (const float*)d_in[2];
    const float* Wa_b  = (const float*)d_in[3];
    const float* Ua_w  = (const float*)d_in[4];
    const float* Ua_b  = (const float*)d_in[5];
    const float* Va_w  = (const float*)d_in[6];
    const float* Va_b  = (const float*)d_in[7];

    float* out     = (float*)d_out;
    float* context = out;                 // B*H floats
    float* weights = out + Bsz * Hsz;     // B*S floats

    char* ws = (char*)d_ws;
    float* qkb               = (float*)ws;                         // 128 KB
    float* scores            = (float*)(ws + 131072);              // 256 KB
    unsigned short* UaT_hi   = (unsigned short*)(ws + 393216);     // 2 MB
    unsigned short* UaT_lo   = (unsigned short*)(ws + 393216 + 2097152); // 2 MB

    uaconv_kernel<<<dim3(Hsz / 32, Hsz / 32), 256, 0, stream>>>(Ua_w, UaT_hi, UaT_lo);
    qproj_kernel<<<dim3(Hsz / 256, Bsz), 256, 0, stream>>>(query, Wa_w, Wa_b, Ua_b, qkb);

    const size_t smemBytes = (size_t)(2 * ROWS * Hsz) * sizeof(unsigned short) +
                             ROWS * sizeof(float);   // 262,400 B <= 320 KB WGP LDS
    scores_kernel<<<dim3(Ssz / ROWS, Bsz), BLOCK, smemBytes, stream>>>(
        keys, UaT_hi, UaT_lo, qkb, Va_w, Va_b, scores);

    softmax_kernel<<<Bsz, 256, 0, stream>>>(scores, weights);

    context_kernel<<<dim3(Hsz / 256, Bsz), 256, 0, stream>>>(weights, keys, context);
}
